// NeighborDecoder_39865886442222
// MI455X (gfx1250) — matrix operand order repaired and checked by reading, not compiled, not run on hardware
//
#include <hip/hip_runtime.h>
#include <hip/hip_bf16.h>
#include <math.h>

#define B_    32
#define C_    384
#define HW_   1024
#define N1_   1025
#define NP_   1040   /* rows padded to 65*16 */
#define NCT_  65
#define T_    48
#define NCLS_ 97

#define INV_SCALE 0.05103103630798288f   /* 1/sqrt(384) */

typedef __bf16        v16bf __attribute__((ext_vector_type(16)));
typedef float         v8f   __attribute__((ext_vector_type(8)));
typedef unsigned int  u32x4 __attribute__((ext_vector_type(4)));
typedef int           i32x8 __attribute__((ext_vector_type(8)));
typedef int           i32x4 __attribute__((ext_vector_type(4)));

union Frag { u32x4 q[2]; v16bf b; unsigned short u[16]; };

#if defined(__has_builtin)
#if __has_builtin(__builtin_amdgcn_tensor_load_to_lds) && \
    __has_builtin(__builtin_amdgcn_s_wait_tensorcnt)
#define HAVE_TDM 1
#endif
#endif

__device__ __forceinline__ v8f wmma_bf16(v16bf a, v16bf b, v8f c) {
  // D(f32 16x16) = A(bf16 16x32) * B(bf16 32x16) + C
  return __builtin_amdgcn_wmma_f32_16x16x32_bf16(false, a, false, b,
                                                 (short)0, c, false, false);
}

// A-fragment: per-lane two 8-element K chunks (ISA 7.12.2, 16-bit A 16x32)
__device__ __forceinline__ v16bf frag_a(const __bf16* row, int ka) {
  Frag f;
  f.q[0] = *(const u32x4*)(row + ka);
  f.q[1] = *(const u32x4*)(row + ka + 16);
  return f.b;
}
// B-fragment: per-lane two 8-element K chunks (ISA 7.12.2, 16-bit B 32x16)
__device__ __forceinline__ v16bf frag_b(const __bf16* row, int kb) {
  Frag f;
  f.q[0] = *(const u32x4*)(row + kb);
  f.q[1] = *(const u32x4*)(row + kb + 16);
  return f.b;
}

// ---------------------------------------------------------------------------
// 1) Weight prep (bf16): wh rows 0..383 = k_w, rows 384..767 = W2t where
//    W2t[oc][ch] = sum_m q_w[m][ch] * nav_w[m][oc];  b2 = q_b @ nav_w (f32)
// ---------------------------------------------------------------------------
__global__ void k_fold(const float* __restrict__ qw, const float* __restrict__ navw,
                       const float* __restrict__ qb, const float* __restrict__ kw,
                       __bf16* __restrict__ wh, float* __restrict__ b2) {
  int idx = blockIdx.x * blockDim.x + threadIdx.x;
  if (idx < 768 * C_) {
    int oc = idx / C_, ch = idx % C_;
    if (oc < C_) {
      wh[idx] = (__bf16)kw[oc * C_ + ch];
    } else {
      int o2 = oc - C_;
      float s = 0.f;
      for (int m = 0; m < C_; ++m) s += qw[m * C_ + ch] * navw[m * C_ + o2];
      wh[idx] = (__bf16)s;
    }
  }
  if (idx < C_) {
    float s = 0.f;
    for (int m = 0; m < C_; ++m) s += qb[m] * navw[m * C_ + idx];
    b2[idx] = s;
  }
}

// ---------------------------------------------------------------------------
// 2) g[b][ch] = mean_n x[b][ch][n]   (one wave per row)
// ---------------------------------------------------------------------------
__global__ void k_mean(const float* __restrict__ x, float* __restrict__ g) {
  int t = threadIdx.x, w = t >> 5, lane = t & 31;
  int row = blockIdx.x * 8 + w;
  const float* pr = x + (size_t)row * HW_;
  float s = 0.f;
  for (int k = lane; k < HW_; k += 32) s += pr[k];
  for (int off = 16; off > 0; off >>= 1) s += __shfl_xor(s, off, 32);
  if (lane == 0) g[row] = s * (1.0f / HW_);
}

// ---------------------------------------------------------------------------
// 3) q_start[b][oc] = g[b] . q_w[oc] + q_b[oc]
// ---------------------------------------------------------------------------
__global__ void k_qstart(const float* __restrict__ g, const float* __restrict__ qw,
                         const float* __restrict__ qb, float* __restrict__ qs) {
  int idx = blockIdx.x * blockDim.x + threadIdx.x;
  int b = idx / C_, oc = idx % C_;
  float s = qb[oc];
  const float* gr = g + (size_t)b * C_;
  const float* wr = qw + (size_t)oc * C_;
  for (int ch = 0; ch < C_; ++ch) s += gr[ch] * wr[ch];
  qs[idx] = s;
}

// ---------------------------------------------------------------------------
// 4) WMMA GEMM (A staged via TDM): per (b, 16-row tile of x_ext)
//      kfh[b][row][:] = bf16( x_ext_row @ k_w^T + k_b )   (rows>1024 zeroed)
//      ph [b][row][:] = bf16( x_ext_row @ W2t^T + b2  )   (rows<1024 only)
// ---------------------------------------------------------------------------
__global__ void __launch_bounds__(128)
k_featgemm(const float* __restrict__ x, const float* __restrict__ eos,
           const __bf16* __restrict__ wh, const float* __restrict__ kb,
           const float* __restrict__ b2,
           __bf16* __restrict__ kfh, __bf16* __restrict__ ph) {
  int rt = blockIdx.x, b = blockIdx.y;
  __shared__ float  Xs[16 * C_];    // raw f32 tile (TDM dest), [ch][16 n]
  __shared__ __bf16 As[16 * C_];    // bf16 A tile, [m][ch]
  int t = threadIdx.x;

#if HAVE_TDM
  if (rt < 64) {
    if (t < 32) {   // one TDM issue per workgroup (wave 0)
      unsigned lds_byte = (unsigned)(size_t)(void*)&Xs[0];
      unsigned long long ga =
          (unsigned long long)(const void*)(x + ((size_t)b * C_) * HW_ + rt * 16);
      u32x4 g0 = { 1u,                                  // count=1 (valid D#)
                   lds_byte,                            // lds_addr
                   (unsigned)ga,                        // global_addr[31:0]
                   (unsigned)(ga >> 32) | (2u << 30) }; // addr[56:32] | type=2
      i32x8 g1 = { (int)(2u << 16),          // data_size = 4 bytes
                   (int)(HW_ << 16),         // tensor_dim0[15:0] in [31:16]
                   (int)(C_ << 16),          // dim0 hi | tensor_dim1[15:0]
                   (int)(16 << 16),          // dim1 hi | tile_dim0 = 16
                   (int)C_,                  // tile_dim1 = 384 (tile_dim2=0)
                   (int)HW_,                 // tensor_dim0_stride = 1024
                   0, 0 };
      i32x4 g2 = { 0, 0, 0, 0 };
      i32x4 g3 = { 0, 0, 0, 0 };
      i32x8 gz = { 0, 0, 0, 0, 0, 0, 0, 0 };
      __builtin_amdgcn_tensor_load_to_lds(g0, g1, g2, g3, gz, 0);
      __builtin_amdgcn_s_wait_tensorcnt(0);
    }
  }
#else
  if (rt < 64) {
    for (int idx = t; idx < 16 * C_; idx += 128) {
      int ch = idx >> 4, m = idx & 15;
      Xs[ch * 16 + m] = x[((size_t)b * C_ + ch) * HW_ + rt * 16 + m];
    }
  }
#endif
  __syncthreads();
  // convert/patch into bf16 A tile [m][ch]
  for (int idx = t; idx < 16 * C_; idx += 128) {
    int ch = idx >> 4, m = idx & 15;
    float v;
    if (rt < 64)      v = Xs[ch * 16 + m];
    else if (m == 0)  v = eos[ch];          // row 1024 = eos
    else              v = 0.f;              // pad rows
    As[m * C_ + ch] = (__bf16)v;
  }
  __syncthreads();

  int wave = t >> 5, lane = t & 31;
  int mrow = lane & 15;
  int hi = (lane & 16) ? 1 : 0;
  const __bf16* arow = &As[mrow * C_];

  for (int ct = wave; ct < 48; ct += 4) {            // 768 output cols
    v8f acc = {0.f, 0.f, 0.f, 0.f, 0.f, 0.f, 0.f, 0.f};
    const __bf16* brow = wh + (size_t)(ct * 16 + mrow) * C_;
    for (int ks = 0; ks < 12; ++ks) {
      __builtin_prefetch(brow + ks * 32 + 64, 0, 1);
      acc = wmma_bf16(frag_a(arow, ks * 32 + hi * 8),
                      frag_b(brow, ks * 32 + hi * 8), acc);
    }
    int rbase = hi * 8;
    if (ct < 24) {
      int oc = ct * 16 + mrow;
      float bias = kb[oc];
#pragma unroll
      for (int r = 0; r < 8; ++r) {
        int row = rt * 16 + rbase + r;
        float v = (row <= HW_) ? (acc[r] + bias) : 0.f;
        kfh[((size_t)b * NP_ + row) * C_ + oc] = (__bf16)v;
      }
    } else {
      int oc = (ct - 24) * 16 + mrow;
      float bias = b2[oc];
#pragma unroll
      for (int r = 0; r < 8; ++r) {
        int row = rt * 16 + rbase + r;
        if (row < HW_)
          ph[((size_t)b * HW_ + row) * C_ + oc] = (__bf16)(acc[r] + bias);
      }
    }
  }
}

// ---------------------------------------------------------------------------
// 5) WMMA GEMM + fused row softmax (pure b128 loads + wmma inner loop):
//      nb[b][m][n] = softmax_n( (p[b][m] . kf[b][n] + nav_b) / sqrt(c) )
// ---------------------------------------------------------------------------
__global__ void __launch_bounds__(128)
k_nbgemm(const __bf16* __restrict__ ph, const __bf16* __restrict__ kfh,
         const float* __restrict__ navb, float* __restrict__ nb) {
  int rt = blockIdx.x, b = blockIdx.y;
  int t = threadIdx.x, wave = t >> 5, lane = t & 31;
  int mrow = lane & 15;
  int hi = (lane & 16) ? 1 : 0;
  int rbase = hi * 8, cin = mrow;
  float navb0 = navb[0];

  const __bf16* Ar = ph + ((size_t)b * HW_ + rt * 16 + mrow) * C_;
  const __bf16* Bb = kfh + (size_t)b * NP_ * C_;

  v8f acc[17];
#pragma unroll
  for (int i = 0; i < 17; ++i)
    acc[i] = (v8f){0.f, 0.f, 0.f, 0.f, 0.f, 0.f, 0.f, 0.f};

#pragma unroll
  for (int i = 0; i < 17; ++i) {
    int ct = wave + 4 * i;
    if (ct < NCT_) {
      const __bf16* brow = Bb + (size_t)(ct * 16 + mrow) * C_;
      v8f a = acc[i];
      for (int ks = 0; ks < 12; ++ks) {
        __builtin_prefetch(brow + ks * 32 + 64, 0, 1);
        a = wmma_bf16(frag_a(Ar, ks * 32 + hi * 8),
                      frag_b(brow, ks * 32 + hi * 8), a);
      }
      acc[i] = a;
    }
  }

  __shared__ float wmax[4][16], wsum[4][16], rmax[16], rsum[16];

  // pass 1: row max (cols 0..1024 only)
  float lmax[8];
#pragma unroll
  for (int r = 0; r < 8; ++r) lmax[r] = -3.0e38f;
#pragma unroll
  for (int i = 0; i < 17; ++i) {
    int ct = wave + 4 * i;
    if (ct < NCT_ && (ct * 16 + cin) <= HW_) {
#pragma unroll
      for (int r = 0; r < 8; ++r)
        lmax[r] = fmaxf(lmax[r], (acc[i][r] + navb0) * INV_SCALE);
    }
  }
#pragma unroll
  for (int m = 1; m < 16; m <<= 1)
#pragma unroll
    for (int r = 0; r < 8; ++r)
      lmax[r] = fmaxf(lmax[r], __shfl_xor(lmax[r], m, 32));
  if (cin == 0)
#pragma unroll
    for (int r = 0; r < 8; ++r) wmax[wave][rbase + r] = lmax[r];
  __syncthreads();
  if (t < 16) {
    float m = -3.0e38f;
    for (int w = 0; w < 4; ++w) m = fmaxf(m, wmax[w][t]);
    rmax[t] = m;
  }
  __syncthreads();
  float rm[8];
#pragma unroll
  for (int r = 0; r < 8; ++r) rm[r] = rmax[rbase + r];

  // pass 2: row sum of exp
  float lsum[8];
#pragma unroll
  for (int r = 0; r < 8; ++r) lsum[r] = 0.f;
#pragma unroll
  for (int i = 0; i < 17; ++i) {
    int ct = wave + 4 * i;
    if (ct < NCT_ && (ct * 16 + cin) <= HW_) {
#pragma unroll
      for (int r = 0; r < 8; ++r)
        lsum[r] += expf((acc[i][r] + navb0) * INV_SCALE - rm[r]);
    }
  }
#pragma unroll
  for (int m = 1; m < 16; m <<= 1)
#pragma unroll
    for (int r = 0; r < 8; ++r) lsum[r] += __shfl_xor(lsum[r], m, 32);
  if (cin == 0)
#pragma unroll
    for (int r = 0; r < 8; ++r) wsum[wave][rbase + r] = lsum[r];
  __syncthreads();
  if (t < 16) {
    float s = 0.f;
    for (int w = 0; w < 4; ++w) s += wsum[w][t];
    rsum[t] = s;
  }
  __syncthreads();
  float rs[8];
#pragma unroll
  for (int r = 0; r < 8; ++r) rs[r] = 1.0f / rsum[rbase + r];

  // pass 3: normalize + store
#pragma unroll
  for (int i = 0; i < 17; ++i) {
    int ct = wave + 4 * i;
    if (ct < NCT_) {
      int col = ct * 16 + cin;
      if (col <= HW_) {
#pragma unroll
        for (int r = 0; r < 8; ++r) {
          int row = rt * 16 + rbase + r;
          float v = expf((acc[i][r] + navb0) * INV_SCALE - rm[r]) * rs[r];
          nb[((size_t)b * N1_ + row) * N1_ + col] = v;
        }
      }
    }
  }
}

// ---------------------------------------------------------------------------
// 6) nb_map last row = one-hot(eos)
// ---------------------------------------------------------------------------
__global__ void k_lastrow(float* __restrict__ nb) {
  int b = blockIdx.x;
  for (int n = threadIdx.x; n < N1_; n += blockDim.x)
    nb[((size_t)b * N1_ + HW_) * N1_ + n] = (n == HW_) ? 1.f : 0.f;
}

// ---------------------------------------------------------------------------
// 7) start_map[b] = softmax_n( kf[b][n] . q_start[b] / sqrt(c) )
// ---------------------------------------------------------------------------
__global__ void k_start(const __bf16* __restrict__ kfh, const float* __restrict__ qs,
                        float* __restrict__ st) {
  int b = blockIdx.x, t = threadIdx.x;
  __shared__ float qv[C_];
  __shared__ float red[256];
  for (int ch = t; ch < C_; ch += 256) qv[ch] = qs[(size_t)b * C_ + ch];
  __syncthreads();
  float l[5];
  int cnt = 0;
  float lm = -3.0e38f;
  for (int n = t; n < N1_; n += 256) {
    const __bf16* kr = kfh + ((size_t)b * NP_ + n) * C_;
    float s = 0.f;
    for (int ch = 0; ch < C_; ++ch) s += (float)kr[ch] * qv[ch];
    s *= INV_SCALE;
    l[cnt++] = s;
    lm = fmaxf(lm, s);
  }
  red[t] = lm; __syncthreads();
  for (int s = 128; s > 0; s >>= 1) { if (t < s) red[t] = fmaxf(red[t], red[t + s]); __syncthreads(); }
  float gmax = red[0]; __syncthreads();
  float ss = 0.f;
  for (int i = 0; i < cnt; ++i) ss += expf(l[i] - gmax);
  red[t] = ss; __syncthreads();
  for (int s = 128; s > 0; s >>= 1) { if (t < s) red[t] += red[t + s]; __syncthreads(); }
  float ginv = 1.0f / red[0];
  cnt = 0;
  for (int n = t; n < N1_; n += 256) st[(size_t)b * N1_ + n] = expf(l[cnt++] - gmax) * ginv;
}

// ---------------------------------------------------------------------------
// 8) align_chars recurrence: one 1024-thread WG per batch; nb_map slice
//    (4.2 MB/batch, 134 MB total) stays resident in the 192 MB L2.
// ---------------------------------------------------------------------------
__global__ void __launch_bounds__(1024)
k_align(const float* __restrict__ st, const float* __restrict__ nb,
        float* __restrict__ maps, float* __restrict__ masks) {
  int b = blockIdx.x, t = threadIdx.x;
  __shared__ float cur[N1_];
  __shared__ float cmd[N1_];
  __shared__ float red[1024];
  __shared__ int fin;
  cur[t] = st[(size_t)b * N1_ + t];
  if (t == 0) { cur[HW_] = st[(size_t)b * N1_ + HW_]; fin = 0; }
  __syncthreads();
  const float* nbB = nb + (size_t)b * N1_ * N1_;
  for (int i = 0; i < T_; ++i) {
    maps[((size_t)b * T_ + i) * N1_ + t] = cur[t];          // emit BEFORE update
    if (t == 0) {
      maps[((size_t)b * T_ + i) * N1_ + HW_] = cur[HW_];
      masks[b * T_ + i] = (fin == 0) ? 1.f : 0.f;
      if (cur[HW_] > 0.6f) fin += 1;
    }
    float kk = fminf((float)(1 + 2 * i), 16.f);
    float fx = expm1f(cur[t] * kk);
    float fxe = (t == 0) ? expm1f(cur[HW_] * kk) : 0.f;
    red[t] = fx + fxe;
    __syncthreads();
    for (int s = 512; s > 0; s >>= 1) { if (t < s) red[t] += red[t + s]; __syncthreads(); }
    float inv = 1.0f / red[0];
    __syncthreads();
    cmd[t] = fx * inv;
    if (t == 0) cmd[HW_] = fxe * inv;
    __syncthreads();
    float a = 0.f;
#pragma unroll 4
    for (int n = 0; n <= HW_; ++n) a += cmd[n] * nbB[(size_t)n * N1_ + t];
    red[t] = cmd[t] * nbB[(size_t)t * N1_ + HW_];
    if (t == 0) red[0] += cmd[HW_] * nbB[(size_t)HW_ * N1_ + HW_];
    __syncthreads();
    for (int s = 512; s > 0; s >>= 1) { if (t < s) red[t] += red[t + s]; __syncthreads(); }
    cur[t] = a;
    if (t == 0) cur[HW_] = red[0];
    __syncthreads();
  }
}

// ---------------------------------------------------------------------------
// 9) char_feats[b][t][ch] = mask * sum_n maps[b][t][n] * x_ext[b][n][ch]
// ---------------------------------------------------------------------------
__global__ void k_cfeat(const float* __restrict__ x, const float* __restrict__ eos,
                        const float* __restrict__ maps, const float* __restrict__ masks,
                        float* __restrict__ cf) {
  int ti = blockIdx.x, b = blockIdx.y, t = threadIdx.x;
  __shared__ float mp[N1_];
  for (int n = t; n < N1_; n += 128) mp[n] = maps[((size_t)b * T_ + ti) * N1_ + n];
  __syncthreads();
  float mask = masks[b * T_ + ti];
  for (int ch = t; ch < C_; ch += 128) {
    const float* xr = x + ((size_t)b * C_ + ch) * HW_;
    float s = 0.f;
    for (int n = 0; n < HW_; ++n) s += mp[n] * xr[n];
    s += mp[HW_] * eos[ch];
    cf[((size_t)b * T_ + ti) * C_ + ch] = s * mask;
  }
}

// ---------------------------------------------------------------------------
// 10) logits = char_feats @ cls_w^T + cls_b
// ---------------------------------------------------------------------------
__global__ void k_logits(const float* __restrict__ cf, const float* __restrict__ clsw,
                         const float* __restrict__ clsb, float* __restrict__ out) {
  int idx = blockIdx.x * blockDim.x + threadIdx.x;
  int row = idx / NCLS_, o = idx % NCLS_;
  const float* cr = cf + (size_t)row * C_;
  const float* wr = clsw + (size_t)o * C_;
  float s = clsb[o];
  for (int ch = 0; ch < C_; ++ch) s += cr[ch] * wr[ch];
  out[(size_t)row * NCLS_ + o] = s;
}

// ---------------------------------------------------------------------------
extern "C" void kernel_launch(void* const* d_in, const int* in_sizes, int n_in,
                              void* d_out, int out_size, void* d_ws, size_t ws_size,
                              hipStream_t stream) {
  (void)in_sizes; (void)n_in; (void)out_size; (void)ws_size;
  const float* x    = (const float*)d_in[0];
  const float* eos  = (const float*)d_in[1];
  const float* qw   = (const float*)d_in[2];
  const float* qb   = (const float*)d_in[3];
  const float* kw   = (const float*)d_in[4];
  const float* kb   = (const float*)d_in[5];
  const float* navw = (const float*)d_in[6];
  const float* navb = (const float*)d_in[7];
  const float* clsw = (const float*)d_in[8];
  const float* clsb = (const float*)d_in[9];

  float* out    = (float*)d_out;
  float* logits = out;                                       // 32*48*97
  float* cf     = logits + (size_t)B_ * T_ * NCLS_;          // 32*48*384
  float* maps   = cf + (size_t)B_ * T_ * C_;                 // 32*48*1025
  float* masks  = maps + (size_t)B_ * T_ * N1_;              // 32*48
  float* nb     = masks + (size_t)B_ * T_;                   // 32*1025*1025

  float*  g   = (float*)d_ws;                                // 32*384
  float*  qs  = g + (size_t)B_ * C_;                         // 32*384
  float*  b2  = qs + (size_t)B_ * C_;                        // 384
  float*  st  = b2 + C_;                                     // 32*1025
  __bf16* wh  = (__bf16*)(st + (size_t)B_ * N1_);            // 768*384 bf16
  __bf16* kfh = wh + (size_t)768 * C_;                       // 32*1040*384 bf16
  __bf16* ph  = kfh + (size_t)B_ * NP_ * C_;                 // 32*1024*384 bf16

  k_fold<<<1152, 256, 0, stream>>>(qw, navw, qb, kw, wh, b2);
  k_mean<<<1536, 256, 0, stream>>>(x, g);
  k_qstart<<<48, 256, 0, stream>>>(g, qw, qb, qs);
  k_featgemm<<<dim3(NCT_, B_), 128, 0, stream>>>(x, eos, wh, kb, b2, kfh, ph);
  k_nbgemm<<<dim3(64, B_), 128, 0, stream>>>(ph, kfh, navb, nb);
  k_lastrow<<<B_, 256, 0, stream>>>(nb);
  k_start<<<B_, 256, 0, stream>>>(kfh, qs, st);
  k_align<<<B_, 1024, 0, stream>>>(st, nb, maps, masks);
  k_cfeat<<<dim3(T_, B_), 128, 0, stream>>>(x, eos, maps, masks, cf);
  k_logits<<<582, 256, 0, stream>>>(cf, clsw, clsb, logits);
}
